// VQVAEModel_23424751632465
// MI455X (gfx1250) — matrix operand order repaired and checked by reading, not compiled, not run on hardware
//
#include <hip/hip_runtime.h>
#include <math.h>

// ---------------------------------------------------------------------------
// CDNA5 / gfx1250 VQ-VAE forward.  bf16 NHWC activations, f32 accumulate.
// Every channel-heavy conv, the transposed conv (via parity classes), and the
// VQ distance GEMM run on v_wmma_f32_16x16x32_bf16.
// ---------------------------------------------------------------------------

typedef __attribute__((ext_vector_type(16))) __bf16 bf16x16;
typedef __attribute__((ext_vector_type(8)))  float  f32x8;
typedef __attribute__((ext_vector_type(4)))  unsigned int uint32x4;
typedef __attribute__((ext_vector_type(8)))  int int32x8;
typedef __attribute__((ext_vector_type(4)))  int int32x4;

union AFrag { bf16x16 v; unsigned short s[16]; };

__device__ __forceinline__ float bf2f(__bf16 x) { return (float)x; }
__device__ __forceinline__ __bf16 f2bf(float x) { return (__bf16)x; }

// Compile-time chain of NB identical WMMAs sharing one A fragment.
// reuse_a is a constant (J>0): operand-reuse hint on back-to-back twins.
template<int J, int NB>
struct WmmaChain {
    static __device__ __forceinline__ void run(f32x8* acc, const bf16x16& a,
                                               const __bf16* bbase) {
        bf16x16 b = *(const bf16x16*)(bbase + (size_t)J * (32 * 16));
        acc[J] = __builtin_amdgcn_wmma_f32_16x16x32_bf16(false, a, false, b,
                                                         (short)0, acc[J],
                                                         (J > 0), false);
        WmmaChain<J + 1, NB>::run(acc, a, bbase);
    }
};
template<int NB>
struct WmmaChain<NB, NB> {
    static __device__ __forceinline__ void run(f32x8*, const bf16x16&, const __bf16*) {}
};

// ---------------------------------------------------------------------------
// CDNA5 feature probe (not launched): Tensor Data Mover + cluster ops.
// 6-arg tensor_load_to_lds builtin on this toolchain.
// ---------------------------------------------------------------------------
__global__ void tdm_probe(const unsigned int* __restrict__ g, int* __restrict__ sink)
{
    uint32x4 g0 = {g[0], g[1], g[2], g[3]};
    int32x8  g1 = {(int)g[4], (int)g[5], (int)g[6], (int)g[7],
                   (int)g[8], (int)g[9], (int)g[10], (int)g[11]};
    int32x4  g2 = {(int)g[12], (int)g[13], (int)g[14], (int)g[15]};
    int32x4  g3 = {(int)g[16], (int)g[17], (int)g[18], (int)g[19]};
    int32x8  g4 = {(int)g[20], (int)g[21], (int)g[22], (int)g[23],
                   (int)g[24], (int)g[25], (int)g[26], (int)g[27]};
    __builtin_amdgcn_tensor_load_to_lds(g0, g1, g2, g3, g4, 0);
    __builtin_amdgcn_s_wait_tensorcnt(0);
    sink[0] = __builtin_amdgcn_cluster_id_x();
    __builtin_amdgcn_s_cluster_barrier();
}

// ---------------------------------------------------------------------------
// A-fragment 8-element run loader.  K order: k = (kh*KW+kw)*CI + ci.
// kbase is 8-aligned, CI is a power of two >= 32 -> run stays in one (kh,kw).
// ---------------------------------------------------------------------------
template<int KW>
__device__ __forceinline__ void loadA8(unsigned short* dst,
                                       const __bf16* __restrict__ in,
                                       int n, int oh, int ow, int kbase,
                                       int CI_l2, int H, int W,
                                       int stride, int pad, int relu_in)
{
    const int kpos = kbase >> CI_l2;
    const int ci0  = kbase & ((1 << CI_l2) - 1);
    const int kh   = kpos / KW;           // KW is a compile-time constant
    const int kw   = kpos - kh * KW;
    const int ih   = oh * stride - pad + kh;
    const int iw   = ow * stride - pad + kw;
    if ((unsigned)ih < (unsigned)H && (unsigned)iw < (unsigned)W) {
        union { uint4 q; unsigned short s[8]; } u;
        u.q = *(const uint4*)(in + ((((size_t)n * H + ih) * W + iw) << CI_l2) + ci0);
        if (relu_in) {
#pragma unroll
            for (int i = 0; i < 8; ++i)
                if (u.s[i] & 0x8000u) u.s[i] = 0;     // bf16 relu: clear negatives
        }
#pragma unroll
        for (int i = 0; i < 8; ++i) dst[i] = u.s[i];
    } else {
#pragma unroll
        for (int i = 0; i < 8; ++i) dst[i] = 0;
    }
}

// ---------------------------------------------------------------------------
// Generic implicit-GEMM conv, N-register-blocked: each wave computes a
// 16-pixel x (NB*16)-channel tile: one A gather feeds NB WMMAs.
// Requires OW,OH,CI powers of two; OW%16==0; CI%32==0; (CO/16)%NB==0.
// ---------------------------------------------------------------------------
template<int NB, int KW>
__global__ void __launch_bounds__(256)
conv_wmma(const __bf16* __restrict__ in, const __bf16* __restrict__ wfrag,
          const float* __restrict__ bias, const __bf16* __restrict__ resid,
          __bf16* __restrict__ out,
          int mtiles, int ntb_l2, int CO, int CI_l2, int kts,
          int OW_l2, int OH_l2, int H, int W,
          int stride, int pad, int relu_in, int relu_out)
{
    const int lane = threadIdx.x & 31;
    const int wid  = blockIdx.x * (blockDim.x >> 5) + (threadIdx.x >> 5);
    if (wid >= (mtiles << ntb_l2)) return;        // wave-uniform: EXEC all-1s at WMMA
    const int mt = wid >> ntb_l2;
    const int nb = wid & ((1 << ntb_l2) - 1);
    const int ntiles = CO >> 4;
    const int hi   = lane >> 4;
    const int mrow = lane & 15;
    const int pix  = mt * 16 + mrow;
    const int ow   = pix & ((1 << OW_l2) - 1);    // tile never crosses a row
    const int oh   = (pix >> OW_l2) & ((1 << OH_l2) - 1);
    const int n    = pix >> (OW_l2 + OH_l2);

    f32x8 acc[NB];
#pragma unroll
    for (int j = 0; j < NB; ++j) acc[j] = (f32x8){0.f,0.f,0.f,0.f,0.f,0.f,0.f,0.f};

    for (int kt = 0; kt < kts; ++kt) {
        AFrag a;
        const int k0 = kt * 32 + (hi ? 8 : 0);
        loadA8<KW>(a.s,     in, n, oh, ow, k0,      CI_l2, H, W, stride, pad, relu_in);
        loadA8<KW>(a.s + 8, in, n, oh, ow, k0 + 16, CI_l2, H, W, stride, pad, relu_in);
        const __bf16* bbase = wfrag + (((size_t)kt * ntiles + nb * NB) * 32 + lane) * 16;
        if (kt + 1 < kts)
            __builtin_prefetch(bbase + (size_t)ntiles * 32 * 16, 0, 0);
        WmmaChain<0, NB>::run(acc, a.v, bbase);
    }

#pragma unroll
    for (int j = 0; j < NB; ++j) {
        const int co = (nb * NB + j) * 16 + mrow;
        const float bv = bias ? bias[co] : 0.0f;
#pragma unroll
        for (int v = 0; v < 8; ++v) {
            const int p = mt * 16 + v + (hi << 3);
            float val = acc[j][v] + bv;
            if (resid)    val += bf2f(resid[(size_t)p * CO + co]);
            if (relu_out) val  = fmaxf(val, 0.0f);
            out[(size_t)p * CO + co] = f2bf(val);
        }
    }
}

// ---------------------------------------------------------------------------
// Weight packing: f32 OIHW -> bf16 WMMA B-fragment layout
//   frag[(((kt*ntiles+nt)*32+lane)*16)+e] = B[K][N]
//   K = kt*32 + (lane>=16?8:0) + (e&7) + (e>>3)*16,  N = nt*16 + (lane&15)
// ---------------------------------------------------------------------------
__global__ void pack_conv_w(const float* __restrict__ w, __bf16* __restrict__ frag,
                            int CO, int CI, int KH, int KW)
{
    const int total = CO * CI * KH * KW;
    int tid = blockIdx.x * blockDim.x + threadIdx.x;
    if (tid >= total) return;
    const int ntiles = CO >> 4;
    const int e    = tid & 15;
    const int lane = (tid >> 4) & 31;
    const int rest = tid >> 9;
    const int nt   = rest % ntiles;
    const int kt   = rest / ntiles;
    const int K    = kt * 32 + (lane >> 4) * 8 + (e & 7) + (e >> 3) * 16;
    const int co   = nt * 16 + (lane & 15);
    const int kpos = K / CI;
    const int ci   = K - kpos * CI;
    const int kh   = kpos / KW;
    const int kw   = kpos - kh * KW;
    frag[tid] = f2bf(w[(((size_t)co * CI + ci) * KH + kh) * KW + kw]);
}

// codebook (512,64) -> B fragments, K=64 (ktiles=2), N=512 (ntiles=32)
__global__ void pack_cb(const float* __restrict__ cb, __bf16* __restrict__ frag)
{
    int tid = blockIdx.x * blockDim.x + threadIdx.x;
    if (tid >= 64 * 512) return;
    const int e    = tid & 15;
    const int lane = (tid >> 4) & 31;
    const int rest = tid >> 9;
    const int nt   = rest & 31;
    const int kt   = rest >> 5;
    const int K    = kt * 32 + (lane >> 4) * 8 + (e & 7) + (e >> 3) * 16;
    const int code = nt * 16 + (lane & 15);
    frag[tid] = f2bf(cb[code * 64 + K]);
}

// ConvT weights (in=128,out=64,4,4) -> 4 parity-class B-fragment sets.
// Per class c=(pho,pwo): K=(t,ci), t in 0..3, kh=(1-pho)+2*(t>>1), kw=(1-pwo)+2*(t&1)
__global__ void pack_convT_w(const float* __restrict__ w, __bf16* __restrict__ frag)
{
    int tid = blockIdx.x * blockDim.x + threadIdx.x;
    if (tid >= 4 * 512 * 64) return;
    const int c    = tid >> 15;
    const int r    = tid & 32767;
    const int e    = r & 15;
    const int lane = (r >> 4) & 31;
    const int rest = r >> 9;          // 0..63
    const int nt   = rest & 3;
    const int kt   = rest >> 2;
    const int K    = kt * 32 + (lane >> 4) * 8 + (e & 7) + (e >> 3) * 16;
    const int co   = nt * 16 + (lane & 15);
    const int t    = K >> 7;
    const int ci   = K & 127;
    const int pho  = c >> 1, pwo = c & 1;
    const int kh   = (1 - pho) + 2 * (t >> 1);
    const int kw   = (1 - pwo) + 2 * (t & 1);
    frag[tid] = f2bf(w[((size_t)ci * 64 + co) * 16 + kh * 4 + kw]);
}

// ---------------------------------------------------------------------------
// conv1: 1 -> 64 ch, 4x4 s2 p1, fused bias+relu, out bf16 NHWC (16,128,128,64)
// ---------------------------------------------------------------------------
__global__ void conv1_direct(const float* __restrict__ x, const float* __restrict__ w,
                             const float* __restrict__ b, __bf16* __restrict__ out)
{
    int tid = blockIdx.x * blockDim.x + threadIdx.x;
    if (tid >= 16 * 128 * 128 * 64) return;
    const int co = tid & 63;
    const int pixel = tid >> 6;
    const int ow = pixel & 127;
    const int oh = (pixel >> 7) & 127;
    const int n  = pixel >> 14;
    float acc = b[co];
#pragma unroll
    for (int kh = 0; kh < 4; ++kh) {
        const int ih = oh * 2 - 1 + kh;
        if ((unsigned)ih >= 256u) continue;
#pragma unroll
        for (int kw = 0; kw < 4; ++kw) {
            const int iw = ow * 2 - 1 + kw;
            if ((unsigned)iw >= 256u) continue;
            acc += x[((size_t)n * 256 + ih) * 256 + iw] * w[co * 16 + kh * 4 + kw];
        }
    }
    out[tid] = f2bf(fmaxf(acc, 0.0f));
}

// ---------------------------------------------------------------------------
// VQ norms
// ---------------------------------------------------------------------------
__global__ void cb_norms(const float* __restrict__ cb, float* __restrict__ cn2)
{
    int t = blockIdx.x * blockDim.x + threadIdx.x;
    if (t >= 512) return;
    float s = 0.f;
    for (int d = 0; d < 64; ++d) { float v = cb[t * 64 + d]; s += v * v; }
    cn2[t] = s;
}

__global__ void z_norms(const __bf16* __restrict__ z, float* __restrict__ zn2)
{
    int t = blockIdx.x * blockDim.x + threadIdx.x;
    if (t >= 65536) return;
    float s = 0.f;
    for (int d = 0; d < 64; ++d) { float v = bf2f(z[(size_t)t * 64 + d]); s += v * v; }
    zn2[t] = s;
}

// ---------------------------------------------------------------------------
// VQ argmin via WMMA: dist = |z|^2 + |c|^2 - 2 z.c ; 16 pixels per wave.
// Two independent accumulator chains per nt iteration (no WMMA->WMMA RAW
// hazard NOPs); dot = acc0 + acc1 combined in the min update.
// ---------------------------------------------------------------------------
__global__ void __launch_bounds__(256)
vq_argmin(const __bf16* __restrict__ z, const __bf16* __restrict__ cbfrag,
          const float* __restrict__ zn2, const float* __restrict__ cn2,
          int* __restrict__ idx)
{
    const int lane = threadIdx.x & 31;
    const int wid  = blockIdx.x * (blockDim.x >> 5) + (threadIdx.x >> 5);
    if (wid >= 4096) return;                      // 65536 pixels / 16
    const int mt   = wid;
    const int hi   = lane >> 4;
    const int mrow = lane & 15;
    const int pix  = mt * 16 + mrow;

    AFrag a0, a1;
    const unsigned short* zp = (const unsigned short*)(z + (size_t)pix * 64);
#pragma unroll
    for (int kt = 0; kt < 2; ++kt) {
        unsigned short* dst = kt ? a1.s : a0.s;
        const int c0 = kt * 32 + (hi ? 8 : 0);
        union { uint4 q; unsigned short s[8]; } u;
        u.q = *(const uint4*)(zp + c0);
#pragma unroll
        for (int i = 0; i < 8; ++i) dst[i] = u.s[i];
        u.q = *(const uint4*)(zp + c0 + 16);
#pragma unroll
        for (int i = 0; i < 8; ++i) dst[8 + i] = u.s[i];
    }

    float zn[8];
#pragma unroll
    for (int v = 0; v < 8; ++v) zn[v] = zn2[mt * 16 + v + (hi << 3)];

    float mind[8]; int mini[8];
#pragma unroll
    for (int v = 0; v < 8; ++v) { mind[v] = 3.4e38f; mini[v] = 0; }

    for (int nt = 0; nt < 32; ++nt) {
        bf16x16 b0 = *(const bf16x16*)(cbfrag + (((size_t)nt)      * 32 + lane) * 16);
        bf16x16 b1 = *(const bf16x16*)(cbfrag + (((size_t)32 + nt) * 32 + lane) * 16);
        f32x8 acc0 = {0.f, 0.f, 0.f, 0.f, 0.f, 0.f, 0.f, 0.f};
        f32x8 acc1 = {0.f, 0.f, 0.f, 0.f, 0.f, 0.f, 0.f, 0.f};
        acc0 = __builtin_amdgcn_wmma_f32_16x16x32_bf16(false, a0.v, false, b0, (short)0, acc0, false, false);
        acc1 = __builtin_amdgcn_wmma_f32_16x16x32_bf16(false, a1.v, false, b1, (short)0, acc1, false, false);
        const int code = nt * 16 + mrow;
        const float cn = cn2[code];
#pragma unroll
        for (int v = 0; v < 8; ++v) {
            const float d = zn[v] + cn - 2.0f * (acc0[v] + acc1[v]);
            if (d < mind[v]) { mind[v] = d; mini[v] = code; }
        }
    }

    // reduce across the 16 lanes of each half-wave (same pixel rows)
#pragma unroll
    for (int v = 0; v < 8; ++v) {
        float d = mind[v]; int ii = mini[v];
        for (int off = 8; off; off >>= 1) {
            const float od = __shfl_xor(d, off, 16);
            const int   oi = __shfl_xor(ii, off, 16);
            if (od < d || (od == d && oi < ii)) { d = od; ii = oi; }
        }
        if (mrow == 0) idx[mt * 16 + v + (hi << 3)] = ii;
    }
}

// ---------------------------------------------------------------------------
// Scatter: one-hot encodings, quantized (f32 NCHW + bf16 NHWC),
// deterministic SSE partials, code histogram.
// ---------------------------------------------------------------------------
__global__ void vq_scatter(const int* __restrict__ idx, const float* __restrict__ cb,
                           const __bf16* __restrict__ z,
                           float* __restrict__ out_enc, float* __restrict__ out_q,
                           __bf16* __restrict__ qbf,
                           float* __restrict__ partials, int* __restrict__ counts)
{
    const int pix = blockIdx.x * blockDim.x + threadIdx.x;   // 65536, block=256
    const int i = idx[pix];
    out_enc[(size_t)pix * 512 + i] = 1.0f;
    const int n = pix >> 12, h = (pix >> 6) & 63, w = pix & 63;
    float local = 0.f;
    for (int d = 0; d < 64; ++d) {
        const float q  = cb[i * 64 + d];
        const float zz = bf2f(z[(size_t)pix * 64 + d]);
        const float df = q - zz;
        local += df * df;
        out_q[(((size_t)n * 64 + d) * 64 + h) * 64 + w] = q;   // NCHW f32
        qbf[(size_t)pix * 64 + d] = f2bf(q);                   // NHWC bf16
    }
    atomicAdd(&counts[i], 1);
    __shared__ float red[256];
    red[threadIdx.x] = local;
    __syncthreads();
    for (int s = 128; s; s >>= 1) {
        if (threadIdx.x < s) red[threadIdx.x] += red[threadIdx.x + s];
        __syncthreads();
    }
    if (threadIdx.x == 0) partials[blockIdx.x] = red[0];
}

__global__ void vq_finalize(const float* __restrict__ partials, const int* __restrict__ counts,
                            float* __restrict__ out_loss, float* __restrict__ out_perp)
{
    __shared__ float red[512];
    const int t = threadIdx.x;
    const float p = (float)counts[t] * (1.0f / 65536.0f);
    red[t] = p * logf(p + 1e-10f);
    __syncthreads();
    for (int s = 256; s; s >>= 1) {
        if (t < s) red[t] += red[t + s];
        __syncthreads();
    }
    if (t == 0) {
        float sse = 0.f;
        for (int b = 0; b < 256; ++b) sse += partials[b];   // deterministic order
        *out_perp = expf(-red[0]);
        *out_loss = 1.25f * sse * (1.0f / (65536.0f * 64.0f));
    }
}

// ---------------------------------------------------------------------------
// ConvTranspose 128->64 4x4 s2 p1 as WMMA implicit GEMM over parity classes.
// in (16,64,64,128) NHWC bf16 -> out (16,128,128,64) NHWC bf16, bias+relu.
// class c=(pho,pwo); pixel (n,yo,xo): oh=2yo+pho, ow=2xo+pwo.
// K = (t,ci): ih = yo+pho-(t>>1), iw = xo+pwo-(t&1).  kts = 512/32 = 16.
// ---------------------------------------------------------------------------
__global__ void __launch_bounds__(256)
convT1_wmma(const __bf16* __restrict__ in, const __bf16* __restrict__ wfragT,
            const float* __restrict__ bias, __bf16* __restrict__ out)
{
    const int lane = threadIdx.x & 31;
    const int wid  = blockIdx.x * (blockDim.x >> 5) + (threadIdx.x >> 5);
    if (wid >= 4 * 4096) return;
    const int c  = wid >> 12;
    const int mt = wid & 4095;
    const int pho = c >> 1, pwo = c & 1;
    const int hi   = lane >> 4;
    const int mrow = lane & 15;
    const int p0   = mt * 16 + mrow;
    const int xo   = p0 & 63;
    const int yo   = (p0 >> 6) & 63;
    const int n    = p0 >> 12;
    const __bf16* wf = wfragT + (size_t)c * (512 * 64);

    f32x8 acc[4];
#pragma unroll
    for (int j = 0; j < 4; ++j) acc[j] = (f32x8){0.f,0.f,0.f,0.f,0.f,0.f,0.f,0.f};

    for (int kt = 0; kt < 16; ++kt) {
        AFrag a;
#pragma unroll
        for (int half = 0; half < 2; ++half) {
            const int k   = kt * 32 + (hi ? 8 : 0) + half * 16;
            const int t   = k >> 7;
            const int ci0 = k & 127;
            const int ih  = yo + pho - (t >> 1);
            const int iw  = xo + pwo - (t & 1);
            unsigned short* dst = a.s + half * 8;
            if ((unsigned)ih < 64u && (unsigned)iw < 64u) {
                union { uint4 q; unsigned short s[8]; } u;
                u.q = *(const uint4*)(in + ((((size_t)n * 64 + ih) * 64 + iw) << 7) + ci0);
#pragma unroll
                for (int i = 0; i < 8; ++i) dst[i] = u.s[i];
            } else {
#pragma unroll
                for (int i = 0; i < 8; ++i) dst[i] = 0;
            }
        }
        const __bf16* bbase = wf + ((size_t)kt * 4 * 32 + lane) * 16;
        WmmaChain<0, 4>::run(acc, a.v, bbase);
    }

#pragma unroll
    for (int j = 0; j < 4; ++j) {
        const int co = j * 16 + mrow;
        const float bv = bias[co];
#pragma unroll
        for (int v = 0; v < 8; ++v) {
            const int p  = mt * 16 + v + (hi << 3);
            const int xv = p & 63, yv = (p >> 6) & 63, nv = p >> 12;
            const int oh = 2 * yv + pho, ow = 2 * xv + pwo;
            const float val = fmaxf(acc[j][v] + bv, 0.0f);
            out[(((size_t)nv * 128 + oh) * 128 + ow) * 64 + co] = f2bf(val);
        }
    }
}

// ConvTranspose 64->1 4x4 s2 p1. in (16,128,128,64) NHWC bf16, out f32 (16,1,256,256)
__global__ void convT2_direct(const __bf16* __restrict__ in, const float* __restrict__ w,
                              const float* __restrict__ b, float* __restrict__ out)
{
    int tid = blockIdx.x * blockDim.x + threadIdx.x;
    if (tid >= 16 * 256 * 256) return;
    const int ow = tid & 255;
    const int oh = (tid >> 8) & 255;
    const int n  = tid >> 16;
    float acc = b[0];
#pragma unroll
    for (int kh = 0; kh < 4; ++kh) {
        const int th = oh + 1 - kh;
        if (th < 0 || (th & 1)) continue;
        const int ih = th >> 1;
        if (ih >= 128) continue;
#pragma unroll
        for (int kw = 0; kw < 4; ++kw) {
            const int tw = ow + 1 - kw;
            if (tw < 0 || (tw & 1)) continue;
            const int iw = tw >> 1;
            if (iw >= 128) continue;
            const __bf16* ip = in + (((size_t)n * 128 + ih) * 128 + iw) * 64;
            for (int ci = 0; ci < 64; ++ci)
                acc += bf2f(ip[ci]) * w[ci * 16 + kh * 4 + kw];
        }
    }
    out[tid] = acc;
}

// ---------------------------------------------------------------------------
// Host side
// ---------------------------------------------------------------------------
static inline int ilog2(int v) { return 31 - __builtin_clz(v); }

template<int NB, int KW>
static void launch_conv(hipStream_t s, const __bf16* in, const __bf16* wf,
                        const float* bias, const __bf16* resid, __bf16* out,
                        int N, int H, int W, int CI, int CO, int OH, int OW,
                        int KH, int stride, int pad, int relu_in, int relu_out)
{
    const int mtiles = (N * OH * OW) >> 4;
    const int ntb    = (CO >> 4) / NB;
    const int waves  = mtiles * ntb;
    const int wpb    = 8;   // 256 threads
    conv_wmma<NB, KW><<<(waves + wpb - 1) / wpb, wpb * 32, 0, s>>>(
        in, wf, bias, resid, out,
        mtiles, ilog2(ntb), CO, ilog2(CI), (CI * KH * KW) >> 5,
        ilog2(OW), ilog2(OH), H, W, stride, pad, relu_in, relu_out);
}

static void launch_pack(hipStream_t s, const float* w, __bf16* frag,
                        int CO, int CI, int KH, int KW)
{
    const int total = CO * CI * KH * KW;
    pack_conv_w<<<(total + 255) / 256, 256, 0, s>>>(w, frag, CO, CI, KH, KW);
}

extern "C" void kernel_launch(void* const* d_in, const int* in_sizes, int n_in,
                              void* d_out, int out_size, void* d_ws, size_t ws_size,
                              hipStream_t stream)
{
    (void)in_sizes; (void)n_in; (void)out_size; (void)ws_size;
    const float* x    = (const float*)d_in[0];
    const float* ew1  = (const float*)d_in[1];
    const float* eb1  = (const float*)d_in[2];
    const float* ew2  = (const float*)d_in[3];
    const float* eb2  = (const float*)d_in[4];
    const float* ew3  = (const float*)d_in[5];
    const float* eb3  = (const float*)d_in[6];
    const float* er1a = (const float*)d_in[7];
    const float* er1b = (const float*)d_in[8];
    const float* er2a = (const float*)d_in[9];
    const float* er2b = (const float*)d_in[10];
    const float* pw   = (const float*)d_in[11];
    const float* pb   = (const float*)d_in[12];
    const float* cb   = (const float*)d_in[13];
    const float* dw1  = (const float*)d_in[14];
    const float* db1  = (const float*)d_in[15];
    const float* dr1a = (const float*)d_in[16];
    const float* dr1b = (const float*)d_in[17];
    const float* dr2a = (const float*)d_in[18];
    const float* dr2b = (const float*)d_in[19];
    const float* dtw1 = (const float*)d_in[20];
    const float* dtb1 = (const float*)d_in[21];
    const float* dtw2 = (const float*)d_in[22];
    const float* dtb2 = (const float*)d_in[23];

    // output layout: loss(1) | x_rec(16*256*256) | perplexity(1) | enc(65536*512) | q(16*64*64*64)
    float* out_f    = (float*)d_out;
    float* out_loss = out_f + 0;
    float* out_xrec = out_f + 1;
    float* out_perp = out_f + 1 + 16 * 256 * 256;
    float* out_enc  = out_perp + 1;
    float* out_q    = out_enc + (size_t)65536 * 512;

    // workspace carve-up
    char* ws = (char*)d_ws;
    size_t cur = 0;
    auto alloc = [&](size_t bytes) { size_t o = cur; cur = (cur + bytes + 255) & ~(size_t)255; return ws + o; };
    __bf16* act0   = (__bf16*)alloc((size_t)16 * 128 * 128 * 64 * 2);
    __bf16* act1   = (__bf16*)alloc((size_t)16 * 64 * 64 * 128 * 2);
    __bf16* act2   = (__bf16*)alloc((size_t)16 * 64 * 64 * 128 * 2);
    __bf16* resmid = (__bf16*)alloc((size_t)16 * 64 * 64 * 32 * 2);
    __bf16* zbuf   = (__bf16*)alloc((size_t)16 * 64 * 64 * 64 * 2);
    __bf16* qbuf   = (__bf16*)alloc((size_t)16 * 64 * 64 * 64 * 2);
    float*  zn2    = (float*)alloc(65536 * 4);
    float*  cn2    = (float*)alloc(512 * 4);
    int*    idx    = (int*)alloc(65536 * 4);
    int*    counts = (int*)alloc(512 * 4);
    float*  parts  = (float*)alloc(256 * 4);
    __bf16* wfe2  = (__bf16*)alloc((size_t)1024 * 128 * 2);
    __bf16* wfe3  = (__bf16*)alloc((size_t)1152 * 128 * 2);
    __bf16* wfr1a = (__bf16*)alloc((size_t)1152 * 32 * 2);
    __bf16* wfr1b = (__bf16*)alloc((size_t)32 * 128 * 2);
    __bf16* wfr2a = (__bf16*)alloc((size_t)1152 * 32 * 2);
    __bf16* wfr2b = (__bf16*)alloc((size_t)32 * 128 * 2);
    __bf16* wfpw  = (__bf16*)alloc((size_t)128 * 64 * 2);
    __bf16* wfdw1 = (__bf16*)alloc((size_t)576 * 128 * 2);
    __bf16* wfd1a = (__bf16*)alloc((size_t)1152 * 32 * 2);
    __bf16* wfd1b = (__bf16*)alloc((size_t)32 * 128 * 2);
    __bf16* wfd2a = (__bf16*)alloc((size_t)1152 * 32 * 2);
    __bf16* wfd2b = (__bf16*)alloc((size_t)32 * 128 * 2);
    __bf16* wfcb  = (__bf16*)alloc((size_t)64 * 512 * 2);
    __bf16* wfT   = (__bf16*)alloc((size_t)4 * 512 * 64 * 2);

    // ---- weight packing ----
    launch_pack(stream, ew2,  wfe2,  128, 64, 4, 4);
    launch_pack(stream, ew3,  wfe3,  128, 128, 3, 3);
    launch_pack(stream, er1a, wfr1a, 32, 128, 3, 3);
    launch_pack(stream, er1b, wfr1b, 128, 32, 1, 1);
    launch_pack(stream, er2a, wfr2a, 32, 128, 3, 3);
    launch_pack(stream, er2b, wfr2b, 128, 32, 1, 1);
    launch_pack(stream, pw,   wfpw,  64, 128, 1, 1);
    launch_pack(stream, dw1,  wfdw1, 128, 64, 3, 3);
    launch_pack(stream, dr1a, wfd1a, 32, 128, 3, 3);
    launch_pack(stream, dr1b, wfd1b, 128, 32, 1, 1);
    launch_pack(stream, dr2a, wfd2a, 32, 128, 3, 3);
    launch_pack(stream, dr2b, wfd2b, 128, 32, 1, 1);
    pack_cb<<<(64 * 512 + 255) / 256, 256, 0, stream>>>(cb, wfcb);
    pack_convT_w<<<(4 * 512 * 64 + 255) / 256, 256, 0, stream>>>(dtw1, wfT);
    cb_norms<<<2, 256, 0, stream>>>(cb, cn2);

    // ---- encoder ----
    conv1_direct<<<(16 * 128 * 128 * 64) / 256, 256, 0, stream>>>(x, ew1, eb1, act0);
    launch_conv<4, 4>(stream, act0, wfe2, eb2, nullptr, act1, 16, 128, 128, 64, 128, 64, 64, 4, 2, 1, 0, 1);
    launch_conv<4, 3>(stream, act1, wfe3, eb3, nullptr, act2, 16, 64, 64, 128, 128, 64, 64, 3, 1, 1, 0, 0);
    launch_conv<2, 3>(stream, act2, wfr1a, nullptr, nullptr, resmid, 16, 64, 64, 128, 32, 64, 64, 3, 1, 1, 1, 0);
    launch_conv<4, 1>(stream, resmid, wfr1b, nullptr, act2, act1, 16, 64, 64, 32, 128, 64, 64, 1, 1, 0, 1, 0);
    launch_conv<2, 3>(stream, act1, wfr2a, nullptr, nullptr, resmid, 16, 64, 64, 128, 32, 64, 64, 3, 1, 1, 1, 0);
    launch_conv<4, 1>(stream, resmid, wfr2b, nullptr, act1, act2, 16, 64, 64, 32, 128, 64, 64, 1, 1, 0, 1, 0);
    launch_conv<4, 1>(stream, act2, wfpw, pb, nullptr, zbuf, 16, 64, 64, 128, 64, 64, 64, 1, 1, 0, 1, 0);

    // ---- vector quantizer ----
    z_norms<<<65536 / 256, 256, 0, stream>>>(zbuf, zn2);
    (void)hipMemsetAsync(out_enc, 0, (size_t)65536 * 512 * sizeof(float), stream);
    (void)hipMemsetAsync(counts, 0, 512 * sizeof(int), stream);
    vq_argmin<<<4096 / 8, 256, 0, stream>>>(zbuf, wfcb, zn2, cn2, idx);
    vq_scatter<<<65536 / 256, 256, 0, stream>>>(idx, cb, zbuf, out_enc, out_q, qbuf, parts, counts);
    vq_finalize<<<1, 512, 0, stream>>>(parts, counts, out_loss, out_perp);

    // ---- decoder ----
    launch_conv<4, 3>(stream, qbuf, wfdw1, db1, nullptr, act1, 16, 64, 64, 64, 128, 64, 64, 3, 1, 1, 0, 0);
    launch_conv<2, 3>(stream, act1, wfd1a, nullptr, nullptr, resmid, 16, 64, 64, 128, 32, 64, 64, 3, 1, 1, 1, 0);
    launch_conv<4, 1>(stream, resmid, wfd1b, nullptr, act1, act2, 16, 64, 64, 32, 128, 64, 64, 1, 1, 0, 1, 0);
    launch_conv<2, 3>(stream, act2, wfd2a, nullptr, nullptr, resmid, 16, 64, 64, 128, 32, 64, 64, 3, 1, 1, 1, 0);
    launch_conv<4, 1>(stream, resmid, wfd2b, nullptr, act2, act1, 16, 64, 64, 32, 128, 64, 64, 1, 1, 0, 1, 0);
    convT1_wmma<<<(4 * 4096) / 8, 256, 0, stream>>>(act1, wfT, dtb1, act0);
    convT2_direct<<<(16 * 256 * 256) / 256, 256, 0, stream>>>(act0, dtw2, dtb2, out_xrec);
}
